// GCN_70952859730423
// MI455X (gfx1250) — matrix operand order repaired and checked by reading, not compiled, not run on hardware
//
#include <hip/hip_runtime.h>

typedef __attribute__((ext_vector_type(16))) __bf16 v16bf;
typedef __attribute__((ext_vector_type(8)))  __bf16 v8bf;
typedef __attribute__((ext_vector_type(8)))  float  v8f;

#define BN_EPS 1e-5f

// ---------------- small utility kernels ----------------

__global__ void gcn_zero_f(float* p, long n) {
    long i = (long)blockIdx.x * blockDim.x + threadIdx.x;
    if (i < n) p[i] = 0.0f;
}

__global__ void gcn_deg(const long long* __restrict__ dst, float* __restrict__ deg, int E) {
    int e = blockIdx.x * blockDim.x + threadIdx.x;
    if (e < E) atomicAdd(&deg[(int)dst[e]], 1.0f);
}

__global__ void gcn_dis(float* deg, int n) {
    int i = blockIdx.x * blockDim.x + threadIdx.x;
    if (i < n) deg[i] = rsqrtf(deg[i] + 1.0f);
}

// positional encoder: xyz(4) -> 128 (affine-BN + relu) -> 128 ; one node per block
__global__ void gcn_posenc(const float* __restrict__ x,  const float* __restrict__ w1,
                           const float* __restrict__ b1, const float* __restrict__ gamma,
                           const float* __restrict__ beta, const float* __restrict__ w2,
                           const float* __restrict__ b2, float* __restrict__ pe, int N) {
    __shared__ float e[128];
    int node = blockIdx.x;
    int j = threadIdx.x;                     // 0..127
    const float* xyz = x + (size_t)node * 260 + 256;
    float s = b1[j];
#pragma unroll
    for (int k = 0; k < 4; ++k) s = fmaf(xyz[k], w1[j * 4 + k], s);
    s = s * (gamma[j] * rsqrtf(1.0f + BN_EPS)) + beta[j];
    e[j] = fmaxf(s, 0.0f);
    __syncthreads();
    float acc = b2[j];
    const float* wr = w2 + j * 128;
    for (int k = 0; k < 128; ++k) acc = fmaf(e[k], wr[k], acc);
    pe[(size_t)node * 128 + j] = acc;
}

// feat[n, 0:256] = x[n, 0:256]; feat[n, 256:384] = pe[n] ; output bf16
__global__ void gcn_feat(const float* __restrict__ x, const float* __restrict__ pe,
                         __bf16* __restrict__ feat, long total) {
    long i = (long)blockIdx.x * blockDim.x + threadIdx.x;
    if (i >= total) return;
    long n = i / 384;
    int  c = (int)(i % 384);
    float v = (c < 256) ? x[n * 260 + c] : pe[n * 128 + (c - 256)];
    feat[i] = (__bf16)v;
}

// f32 -> bf16 (optional relu) for next-layer GEMM input
__global__ void gcn_cvt(const float* __restrict__ in, __bf16* __restrict__ out,
                        long total, int relu) {
    long i = (long)blockIdx.x * blockDim.x + threadIdx.x;
    if (i >= total) return;
    float v = in[i];
    if (relu) v = fmaxf(v, 0.0f);
    out[i] = (__bf16)v;
}

// W (K x N, f32, row-major) -> Wt (N x K, bf16, row-major)
__global__ void gcn_wt(const float* __restrict__ W, __bf16* __restrict__ Wt, int K, int N) {
    long i = (long)blockIdx.x * blockDim.x + threadIdx.x;
    long total = (long)K * N;
    if (i >= total) return;
    int n = (int)(i / K);
    int k = (int)(i % K);
    Wt[i] = (__bf16)W[(size_t)k * N + n];
}

// ---------------- WMMA GEMM: C(MxN,f32) = A(MxK,bf16) x Bt(NxK,bf16)^T ----------------
// One wave computes a 16(M) x 64(N) tile: A fragment reused across 4 B fragments.
__global__ void gcn_gemm_wmma(const __bf16* __restrict__ A, const __bf16* __restrict__ Bt,
                              float* __restrict__ C, int M, int N, int K) {
    int wid  = (blockIdx.x * blockDim.x + threadIdx.x) >> 5;
    int lane = threadIdx.x & 31;
    int tilesPerRow = N >> 6;                 // N is a multiple of 64 (256 or 128)
    int rowTile = wid / tilesPerRow;
    int colQ    = wid % tilesPerRow;
    int MTiles  = (M + 15) >> 4;
    if (rowTile >= MTiles) return;

    int row  = rowTile * 16 + (lane & 15);
    int arow = row < M ? row : M - 1;         // clamp for ragged M (M=50000 is exact anyway)
    const __bf16* Ar = A + (size_t)arow * K;
    int aShift = (lane >> 4) * 8;             // ISA 16-bit A 16x32 lane layout
    int bShift = (lane >> 4) * 8;             // ISA 16-bit B 32x16 lane layout
    const __bf16* B0 = Bt + (size_t)(colQ * 64 + (lane & 15)) * K;

    v8f acc0 = {}, acc1 = {}, acc2 = {}, acc3 = {};

    for (int k0 = 0; k0 < K; k0 += 32) {
        // A fragment: K = {k0+aShift .. +7} U {k0+aShift+16 .. +23}
        v8bf alo = *(const v8bf*)(Ar + k0 + aShift);
        v8bf ahi = *(const v8bf*)(Ar + k0 + aShift + 16);
        v16bf a;
#pragma unroll
        for (int i = 0; i < 8; ++i) { a[i] = alo[i]; a[8 + i] = ahi[i]; }
        __builtin_prefetch(Ar + k0 + 32, 0, 0);   // global_prefetch_b8 for next K-slab

#pragma unroll
        for (int q = 0; q < 4; ++q) {
            const __bf16* Bq = B0 + (size_t)q * 16 * K;
            v8bf blo = *(const v8bf*)(Bq + k0 + bShift);
            v8bf bhi = *(const v8bf*)(Bq + k0 + bShift + 16);
            v16bf b;
#pragma unroll
            for (int i = 0; i < 8; ++i) { b[i] = blo[i]; b[8 + i] = bhi[i]; }
            v8f* accp = (q == 0) ? &acc0 : (q == 1) ? &acc1 : (q == 2) ? &acc2 : &acc3;
            *accp = __builtin_amdgcn_wmma_f32_16x16x32_bf16(
                false, a, false, b, (short)0, *accp, false, false);
        }
    }

    // C/D layout: lanes 0-15 -> M = r, lanes 16-31 -> M = r+8 ; N = lane&15 (+ tile base)
    int rbase = rowTile * 16 + ((lane >> 4) ? 8 : 0);
#pragma unroll
    for (int q = 0; q < 4; ++q) {
        const v8f& acc = (q == 0) ? acc0 : (q == 1) ? acc1 : (q == 2) ? acc2 : acc3;
        int n = colQ * 64 + q * 16 + (lane & 15);
#pragma unroll
        for (int r = 0; r < 8; ++r) {
            int rr = rbase + r;
            if (rr < M) C[(size_t)rr * N + n] = acc[r];
        }
    }
}

// agg[n,f] = h[n,f] * dis[n]^2 + bias[f]
__global__ void gcn_agg_init(const float* __restrict__ h, const float* __restrict__ dis,
                             const float* __restrict__ bias, float* __restrict__ agg,
                             long total, int F) {
    long i = (long)blockIdx.x * blockDim.x + threadIdx.x;
    if (i >= total) return;
    long n = i / F;
    int  f = (int)(i % F);
    float d = dis[n];
    agg[i] = fmaf(h[i], d * d, bias[f]);
}

// per-edge scatter: agg[dst] += h[src] * dis[src]*dis[dst] ; one thread per (edge, 4 feats)
__global__ void gcn_scatter(const long long* __restrict__ src, const long long* __restrict__ dst,
                            const float* __restrict__ dis, const float* __restrict__ h,
                            float* __restrict__ agg, int E, int F) {
    long tid = (long)blockIdx.x * blockDim.x + threadIdx.x;
    int chunks = F >> 2;
    long e = tid / chunks;
    if (e >= E) return;
    int c = (int)(tid % chunks) << 2;
    int s = (int)src[e];
    int d = (int)dst[e];
    float w = dis[s] * dis[d];
    float4 hv = *(const float4*)(h + (size_t)s * F + c);
    float* out = agg + (size_t)d * F + c;
    atomicAdd(out + 0, hv.x * w);
    atomicAdd(out + 1, hv.y * w);
    atomicAdd(out + 2, hv.z * w);
    atomicAdd(out + 3, hv.w * w);
}

// ---------------- launcher ----------------

extern "C" void kernel_launch(void* const* d_in, const int* in_sizes, int n_in,
                              void* d_out, int out_size, void* d_ws, size_t ws_size,
                              hipStream_t stream) {
    const float*     x       = (const float*)d_in[0];
    const long long* ei      = (const long long*)d_in[1];
    const float*     pos_w1  = (const float*)d_in[2];
    const float*     pos_b1  = (const float*)d_in[3];
    const float*     bn_g    = (const float*)d_in[4];
    const float*     bn_b    = (const float*)d_in[5];
    const float*     pos_w2  = (const float*)d_in[6];
    const float*     pos_b2  = (const float*)d_in[7];
    const float*     W1      = (const float*)d_in[8];
    const float*     b1      = (const float*)d_in[9];
    const float*     W3      = (const float*)d_in[10];
    const float*     b3      = (const float*)d_in[11];
    const float*     W2      = (const float*)d_in[12];
    const float*     b2      = (const float*)d_in[13];

    const int N = in_sizes[0] / 260;
    const int E = in_sizes[1] / 2;
    const long long* src = ei;
    const long long* dst = ei + E;

    // workspace carve-out (256B aligned)
    char*  ws  = (char*)d_ws;
    size_t off = 0;
    auto carve = [&](size_t bytes) -> char* {
        char* p = ws + off;
        off += (bytes + 255) & ~(size_t)255;
        return p;
    };
    float*  dis    = (float*)carve((size_t)N * 4);
    __bf16* bufA   = (__bf16*)carve((size_t)N * 384 * 2);   // bf16 GEMM inputs
    float*  bufH   = (float*)carve((size_t)N * 256 * 4);    // f32 GEMM outputs (pe aliases here)
    float*  bufAgg = (float*)carve((size_t)N * 256 * 4);    // f32 aggregation
    __bf16* bufWt  = (__bf16*)carve((size_t)384 * 256 * 2); // transposed bf16 weights
    float*  pe     = bufH;                                  // pe consumed before GEMM1 writes bufH

    auto nblk = [](long n, int b) { return (unsigned)((n + b - 1) / b); };

    // degree -> dis
    gcn_zero_f<<<nblk(N, 256), 256, 0, stream>>>(dis, N);
    gcn_deg<<<nblk(E, 256), 256, 0, stream>>>(dst, dis, E);
    gcn_dis<<<nblk(N, 256), 256, 0, stream>>>(dis, N);

    // positional encoder + feature assembly
    gcn_posenc<<<N, 128, 0, stream>>>(x, pos_w1, pos_b1, bn_g, bn_b, pos_w2, pos_b2, pe, N);
    gcn_feat<<<nblk((long)N * 384, 256), 256, 0, stream>>>(x, pe, bufA, (long)N * 384);

    auto run_gemm = [&](const __bf16* A, const __bf16* Bt, float* C, int M, int Nc, int K) {
        long waves   = (long)((M + 15) / 16) * (Nc / 64);
        long threads = waves * 32;
        gcn_gemm_wmma<<<nblk(threads, 256), 256, 0, stream>>>(A, Bt, C, M, Nc, K);
    };

    // ---- layer 1: 384 -> 256 ----
    gcn_wt<<<nblk((long)384 * 256, 256), 256, 0, stream>>>(W1, bufWt, 384, 256);
    run_gemm(bufA, bufWt, bufH, N, 256, 384);
    gcn_agg_init<<<nblk((long)N * 256, 256), 256, 0, stream>>>(bufH, dis, b1, bufAgg, (long)N * 256, 256);
    gcn_scatter<<<nblk((long)E * 64, 256), 256, 0, stream>>>(src, dst, dis, bufH, bufAgg, E, 256);

    // ---- layer 2: 256 -> 256 ----
    gcn_cvt<<<nblk((long)N * 256, 256), 256, 0, stream>>>(bufAgg, bufA, (long)N * 256, 0);
    gcn_wt<<<nblk((long)256 * 256, 256), 256, 0, stream>>>(W3, bufWt, 256, 256);
    run_gemm(bufA, bufWt, bufH, N, 256, 256);
    gcn_agg_init<<<nblk((long)N * 256, 256), 256, 0, stream>>>(bufH, dis, b3, bufAgg, (long)N * 256, 256);
    gcn_scatter<<<nblk((long)E * 64, 256), 256, 0, stream>>>(src, dst, dis, bufH, bufAgg, E, 256);

    // ---- layer 3: relu -> 256 -> 128, straight into d_out ----
    gcn_cvt<<<nblk((long)N * 256, 256), 256, 0, stream>>>(bufAgg, bufA, (long)N * 256, 1);
    gcn_wt<<<nblk((long)256 * 128, 256), 256, 0, stream>>>(W2, bufWt, 256, 128);
    run_gemm(bufA, bufWt, bufH, N, 128, 256);
    float* out = (float*)d_out;
    gcn_agg_init<<<nblk((long)N * 128, 256), 256, 0, stream>>>(bufH, dis, b2, out, (long)N * 128, 128);
    gcn_scatter<<<nblk((long)E * 32, 256), 256, 0, stream>>>(src, dst, dis, bufH, out, E, 128);
}
